// Attention_10900626998039
// MI455X (gfx1250) — matrix operand order repaired and checked by reading, not compiled, not run on hardware
//
#include <hip/hip_runtime.h>
#include <hip/hip_bf16.h>
#include <math.h>
#include <stdint.h>

typedef _Float16 half_t;
typedef __attribute__((ext_vector_type(16))) _Float16 v16h;
typedef __attribute__((ext_vector_type(8)))  _Float16 v8h;
typedef __attribute__((ext_vector_type(4)))  _Float16 v4h;
typedef __attribute__((ext_vector_type(8)))  float    v8f;
typedef __attribute__((ext_vector_type(4)))  float    v4f;

#define DIMK   1024
#define NSEQ   2048
#define BATCH  2
#define NHEAD  16
#define DHEAD  64
#define ROWS   (BATCH*NSEQ)   /* 4096 */

#define PSTR   40   /* P-tile LDS row stride (halves): 80B, 16B aligned, bank-spread */
#define KSTR   72   /* K-tile LDS row stride (halves): 144B */
#define VSTR   40   /* V-tile LDS row stride (halves): 80B */
#define BSTR   40   /* weight-tile LDS row stride (halves) */

// ---------------- helpers ----------------

static __device__ __forceinline__ v16h make_a(const half_t* p0, const half_t* p1) {
  // 16-bit A-fragment: per lane two contiguous 8-half chunks (ISA 7.12.2)
  v8h lo = *(const v8h*)p0;
  v8h hi = *(const v8h*)p1;
  return __builtin_shufflevector(lo, hi, 0,1,2,3,4,5,6,7,8,9,10,11,12,13,14,15);
}

static __device__ __forceinline__ v8f wmma_f16(v16h a, v16h b, v8f c) {
  return __builtin_amdgcn_wmma_f32_16x16x32_f16(false, a, false, b, (short)0, c, false, false);
}

// Async global->LDS copy, 16B per lane, tracked by ASYNCcnt (CDNA5 GLOBAL_LOAD_ASYNC_TO_LDS_B128).
// LDS address VGPR = low 32 bits of the generic shared pointer (aperture truncation rule).
static __device__ __forceinline__ void async_ld16(half_t* lds_dst, const half_t* gsrc) {
  unsigned loff = (unsigned)(uintptr_t)lds_dst;
  asm volatile("global_load_async_to_lds_b128 %0, %1, off"
               :: "v"(loff), "v"(gsrc) : "memory");
}
static __device__ __forceinline__ void wait_async0() {
  asm volatile("s_wait_asynccnt 0" ::: "memory");
}
static __device__ __forceinline__ void wait_ds0() {
  asm volatile("s_wait_dscnt 0" ::: "memory");
}

static __device__ __forceinline__ float redmax16(float v) {
  v = fmaxf(v, __shfl_xor(v, 1, 32));
  v = fmaxf(v, __shfl_xor(v, 2, 32));
  v = fmaxf(v, __shfl_xor(v, 4, 32));
  v = fmaxf(v, __shfl_xor(v, 8, 32));
  return v;
}
static __device__ __forceinline__ float redsum16(float v) {
  v += __shfl_xor(v, 1, 32);
  v += __shfl_xor(v, 2, 32);
  v += __shfl_xor(v, 4, 32);
  v += __shfl_xor(v, 8, 32);
  return v;
}
static __device__ __forceinline__ float redsum32(float v) {
  v += __shfl_xor(v, 16, 32);
  return redsum16(v);
}

// ---------------- K0: weight f32 -> f16 ----------------

__global__ void k_convert_w(const float* __restrict__ wq, const float* __restrict__ wk,
                            const float* __restrict__ wv, const float* __restrict__ wo,
                            half_t* __restrict__ dst) {
  int i = blockIdx.x * blockDim.x + threadIdx.x;   // 0 .. 1M-1
  dst[0*(size_t)DIMK*DIMK + i] = (half_t)wq[i];
  dst[1*(size_t)DIMK*DIMK + i] = (half_t)wk[i];
  dst[2*(size_t)DIMK*DIMK + i] = (half_t)wv[i];
  dst[3*(size_t)DIMK*DIMK + i] = (half_t)wo[i];
}

// ---------------- K1: LayerNorm, one wave per row, output f16 ----------------

__global__ void k_layernorm(const float* __restrict__ x, const float* __restrict__ g,
                            const float* __restrict__ bta, half_t* __restrict__ xh) {
  int row  = (blockIdx.x * blockDim.x + threadIdx.x) >> 5;
  int lane = threadIdx.x & 31;
  if (row >= ROWS) return;
  const float* xr = x + (size_t)row * DIMK;
  float vals[32];
  float s = 0.f, ss = 0.f;
#pragma unroll
  for (int it = 0; it < 8; ++it) {
    v4f v = *(const v4f*)(xr + it*128 + lane*4);
#pragma unroll
    for (int j = 0; j < 4; ++j) { vals[it*4+j] = v[j]; s += v[j]; ss += v[j]*v[j]; }
  }
  s  = redsum32(s);
  ss = redsum32(ss);
  float mu   = s * (1.0f / DIMK);
  float var  = ss * (1.0f / DIMK) - mu * mu;
  float rstd = rsqrtf(var + 1e-5f);
  half_t* orow = xh + (size_t)row * DIMK;
#pragma unroll
  for (int it = 0; it < 8; ++it) {
    int c = it*128 + lane*4;
    v4f gg = *(const v4f*)(g + c);
    v4f bb = *(const v4f*)(bta + c);
    v4h o;
#pragma unroll
    for (int j = 0; j < 4; ++j) o[j] = (half_t)((vals[it*4+j] - mu) * rstd * gg[j] + bb[j]);
    *(v4h*)(orow + c) = o;
  }
}

// ---------------- K2: QKV GEMM + fused per-head RMSNorm ----------------
// Block = 8 waves, all on the same (head, matrix): the 64x32 weight chunk is staged
// once per block into double-buffered LDS with async-to-LDS copies. Each wave owns a
// 32(M) x 64(N) tile: 2 A-frags reuse 4 shared B-frags -> 8 WMMA per K-step.

__global__ void k_qkv(const half_t* __restrict__ xh, const half_t* __restrict__ w16,
                      const float* __restrict__ qg, const float* __restrict__ kg,
                      half_t* __restrict__ qb, half_t* __restrict__ kb,
                      half_t* __restrict__ vT) {
  __shared__ half_t sB[2][64 * BSTR];

  int tid  = threadIdx.x;
  int wid  = tid >> 5;
  int lane = tid & 31, lm = lane & 15, lh = lane >> 4;
  int task  = blockIdx.x * 8 + wid;
  int mt    = task & 127;          // 128 M-tiles of 32 rows
  int rest  = task >> 7;
  int head  = rest & 15;
  int which = rest >> 4;           // 0=Q 1=K 2=V
  int m0 = mt * 32, n0 = head * 64;

  const half_t* W = w16 + (size_t)which * DIMK * DIMK;
  // cooperative stage coords: 64 rows x 4 segs of 8 halves
  int srow = tid >> 2, sseg = tid & 3;
  const half_t* wrow = W + (size_t)(n0 + srow) * DIMK + sseg * 8;

  v8f acc[2][4] = {};
  const half_t* arow0 = xh + (size_t)(m0 +      lm) * DIMK;
  const half_t* arow1 = xh + (size_t)(m0 + 16 + lm) * DIMK;

  async_ld16(&sB[0][srow * BSTR + sseg * 8], wrow);   // prologue: stage k0=0
  wait_async0();
  __syncthreads();

  int cur = 0;
  for (int k0 = 0; k0 < DIMK; k0 += 32) {
    if (k0 + 32 < DIMK)
      async_ld16(&sB[cur ^ 1][srow * BSTR + sseg * 8], wrow + k0 + 32);

    v16h a0 = make_a(arow0 + k0 + lh*16, arow0 + k0 + lh*16 + 8);
    v16h a1 = make_a(arow1 + k0 + lh*16, arow1 + k0 + lh*16 + 8);
    const half_t* Bc = sB[cur] + lm * BSTR + lh*16;
    // hoist all four B fragments so the ds_load_b128s overlap (one wait, then WMMA burst)
    v16h b0 = *(const v16h*)(Bc);
    v16h b1 = *(const v16h*)(Bc + 16 * BSTR);
    v16h b2 = *(const v16h*)(Bc + 32 * BSTR);
    v16h b3 = *(const v16h*)(Bc + 48 * BSTR);
    acc[0][0] = wmma_f16(a0, b0, acc[0][0]);
    acc[1][0] = wmma_f16(a1, b0, acc[1][0]);
    acc[0][1] = wmma_f16(a0, b1, acc[0][1]);
    acc[1][1] = wmma_f16(a1, b1, acc[1][1]);
    acc[0][2] = wmma_f16(a0, b2, acc[0][2]);
    acc[1][2] = wmma_f16(a1, b2, acc[1][2]);
    acc[0][3] = wmma_f16(a0, b3, acc[0][3]);
    acc[1][3] = wmma_f16(a1, b3, acc[1][3]);
    wait_async0();
    __syncthreads();
    cur ^= 1;
  }

  if (which < 2) {
    const float* gamma = (which == 0) ? qg : kg;
    float gv[4];
#pragma unroll
    for (int f = 0; f < 4; ++f) gv[f] = gamma[f*16 + lm];
    half_t* outb = (which == 0) ? qb : kb;
#pragma unroll
    for (int s = 0; s < 2; ++s) {
#pragma unroll
      for (int r = 0; r < 8; ++r) {
        float ssq = 0.f;
#pragma unroll
        for (int f = 0; f < 4; ++f) ssq += acc[s][f][r] * acc[s][f][r];
        ssq = redsum16(ssq);                     // row sum of squares over d=64
        float norm = sqrtf(ssq) * 0.125f;        // * d^-0.5
        float sc   = 1.0f / fmaxf(norm, 1e-8f);
        if (which == 0) sc *= 0.125f;            // SCALE
        int grow = m0 + s*16 + r + 8*lh;
        int b = grow >> 11, nn = grow & 2047;
        half_t* dst = outb + ((size_t)(b*NHEAD + head) * NSEQ + nn) * DHEAD;
#pragma unroll
        for (int f = 0; f < 4; ++f) dst[f*16 + lm] = (half_t)(acc[s][f][r] * sc * gv[f]);
      }
    }
  } else {
#pragma unroll
    for (int s = 0; s < 2; ++s) {
#pragma unroll
      for (int r = 0; r < 8; ++r) {
        int grow = m0 + s*16 + r + 8*lh;
        int b = grow >> 11, nn = grow & 2047;
        size_t base = (size_t)(b*NHEAD + head) * DHEAD * NSEQ;
#pragma unroll
        for (int f = 0; f < 4; ++f)
          vT[base + (size_t)(f*16 + lm) * NSEQ + nn] = (half_t)acc[s][f][r];
      }
    }
  }
}

// ---------------- K3: flash attention ----------------
// Block = 8 waves, all on the same (b,h): K (32x64) and V^T (64x32) chunks are staged
// once per block via async-to-LDS (double-buffered); each wave owns 16 query rows.

__global__ void k_attn(const half_t* __restrict__ qb, const half_t* __restrict__ kb,
                       const half_t* __restrict__ vT, half_t* __restrict__ attn) {
  __shared__ half_t sK[2][32 * KSTR];
  __shared__ half_t sV[2][64 * VSTR];
  __shared__ half_t sP[8][16 * PSTR];

  int tid  = threadIdx.x;
  int wid  = tid >> 5;
  int lane = tid & 31, lm = lane & 15, lh = lane >> 4;
  int task = blockIdx.x * 8 + wid;
  int qt = task & 127;            // 128 q-tiles of 16 (8 consecutive -> same bh)
  int bh = task >> 7;             // 0..31
  int qn0 = qt * 16;

  const half_t* kbase = kb + (size_t)bh * NSEQ * DHEAD;
  const half_t* vbase = vT + (size_t)bh * DHEAD * NSEQ;

  // cooperative stage coords
  int kjj = tid >> 3, kseg = tid & 7;          // K: 32 rows x 8 segs
  int vd  = tid >> 2, vseg = tid & 3;          // V: 64 rows x 4 segs
  const half_t* kg = kbase + (size_t)kjj * DHEAD + kseg * 8;
  const half_t* vg = vbase + (size_t)vd * NSEQ + vseg * 8;

  async_ld16(&sK[0][kjj * KSTR + kseg * 8], kg);            // stage chunk 0
  async_ld16(&sV[0][vd  * VSTR + vseg * 8], vg);

  const half_t* qrow = qb + ((size_t)bh * NSEQ + qn0 + lm) * DHEAD;
  v16h aq0 = make_a(qrow      + lh*16, qrow +      lh*16 + 8);    // d 0..31
  v16h aq1 = make_a(qrow + 32 + lh*16, qrow + 32 + lh*16 + 8);    // d 32..63

  v8f o0 = {}, o1 = {}, o2 = {}, o3 = {};
  float mrow[8], lrow[8];
#pragma unroll
  for (int r = 0; r < 8; ++r) { mrow[r] = -__builtin_inff(); lrow[r] = 0.f; }

  half_t* pl = sP[wid];
  wait_async0();
  __syncthreads();

  int cur = 0;
  for (int j0 = 0; j0 < NSEQ; j0 += 32) {
    if (j0 + 32 < NSEQ) {
      async_ld16(&sK[cur ^ 1][kjj * KSTR + kseg * 8], kg + (size_t)32 * DHEAD + j0 * DHEAD);
      async_ld16(&sV[cur ^ 1][vd  * VSTR + vseg * 8], vg + j0 + 32);
    }
    // ---- S = Q K^T from LDS K-tile (fragments hoisted, one wait) ----
    const half_t* Kc = sK[cur] + lm * KSTR + lh*16;
    v16h b00 = *(const v16h*)(Kc);
    v16h b01 = *(const v16h*)(Kc + 32);
    v16h b10 = *(const v16h*)(Kc + 16 * KSTR);
    v16h b11 = *(const v16h*)(Kc + 16 * KSTR + 32);
    v8f s0 = {}, s1 = {};
    s0 = wmma_f16(aq0, b00, s0); s0 = wmma_f16(aq1, b01, s0);
    s1 = wmma_f16(aq0, b10, s1); s1 = wmma_f16(aq1, b11, s1);

    // ---- online softmax (row m = r + 8*lh) ----
    float arow[8];
#pragma unroll
    for (int r = 0; r < 8; ++r) {
      float cm = redmax16(fmaxf(s0[r], s1[r]));
      float mn = fmaxf(mrow[r], cm);
      float al = __expf(mrow[r] - mn);
      mrow[r] = mn; arow[r] = al;
      float p0 = __expf(s0[r] - mn);
      float p1 = __expf(s1[r] - mn);
      lrow[r] = lrow[r] * al + redsum16(p0 + p1);
      s0[r] = p0; s1[r] = p1;
    }
#pragma unroll
    for (int r = 0; r < 8; ++r) { o0[r]*=arow[r]; o1[r]*=arow[r]; o2[r]*=arow[r]; o3[r]*=arow[r]; }

    // ---- stage P (D-fragment) -> LDS -> A-fragment ----
#pragma unroll
    for (int r = 0; r < 8; ++r) {
      pl[(r + 8*lh) * PSTR +      lm] = (half_t)s0[r];
      pl[(r + 8*lh) * PSTR + 16 + lm] = (half_t)s1[r];
    }
    wait_ds0();
    v16h pa = make_a(pl + lm*PSTR + lh*16, pl + lm*PSTR + lh*16 + 8);

    // ---- O += P V from LDS V-tile (fragments hoisted) ----
    const half_t* Vc = sV[cur] + lm * VSTR + lh*16;
    v16h v0 = *(const v16h*)(Vc);
    v16h v1 = *(const v16h*)(Vc + 16 * VSTR);
    v16h v2 = *(const v16h*)(Vc + 32 * VSTR);
    v16h v3 = *(const v16h*)(Vc + 48 * VSTR);
    o0 = wmma_f16(pa, v0, o0);
    o1 = wmma_f16(pa, v1, o1);
    o2 = wmma_f16(pa, v2, o2);
    o3 = wmma_f16(pa, v3, o3);

    wait_async0();
    __syncthreads();
    cur ^= 1;
  }

  // ---- epilogue: normalize, store [b, n, h*64] f16 ----
  int b = bh >> 4, head = bh & 15;
#pragma unroll
  for (int r = 0; r < 8; ++r) {
    float inv = 1.0f / lrow[r];
    int qn = qn0 + r + 8*lh;
    half_t* dst = attn + ((size_t)(b*NSEQ + qn)) * DIMK + head * DHEAD;
    dst[     lm] = (half_t)(o0[r] * inv);
    dst[16 + lm] = (half_t)(o1[r] * inv);
    dst[32 + lm] = (half_t)(o2[r] * inv);
    dst[48 + lm] = (half_t)(o3[r] * inv);
  }
}

// ---------------- K4: out = attn @ Wo.T + bo  (f32 output) ----------------
// Same cooperative LDS staging as K2; M=32 per wave.

__global__ void k_outproj(const half_t* __restrict__ attn, const half_t* __restrict__ wo16,
                          const float* __restrict__ bo, float* __restrict__ out) {
  __shared__ half_t sB[2][64 * BSTR];

  int tid  = threadIdx.x;
  int wid  = tid >> 5;
  int lane = tid & 31, lm = lane & 15, lh = lane >> 4;
  int task = blockIdx.x * 8 + wid;
  int mt = task & 127;
  int nt = task >> 7;              // 16 N-tiles of 64 (uniform per block)
  int m0 = mt * 32, n0 = nt * 64;

  int srow = tid >> 2, sseg = tid & 3;
  const half_t* wrow = wo16 + (size_t)(n0 + srow) * DIMK + sseg * 8;

  v8f acc[2][4] = {};
  const half_t* arow0 = attn + (size_t)(m0 +      lm) * DIMK;
  const half_t* arow1 = attn + (size_t)(m0 + 16 + lm) * DIMK;

  async_ld16(&sB[0][srow * BSTR + sseg * 8], wrow);
  wait_async0();
  __syncthreads();

  int cur = 0;
  for (int k0 = 0; k0 < DIMK; k0 += 32) {
    if (k0 + 32 < DIMK)
      async_ld16(&sB[cur ^ 1][srow * BSTR + sseg * 8], wrow + k0 + 32);

    v16h a0 = make_a(arow0 + k0 + lh*16, arow0 + k0 + lh*16 + 8);
    v16h a1 = make_a(arow1 + k0 + lh*16, arow1 + k0 + lh*16 + 8);
    const half_t* Bc = sB[cur] + lm * BSTR + lh*16;
    v16h b0 = *(const v16h*)(Bc);
    v16h b1 = *(const v16h*)(Bc + 16 * BSTR);
    v16h b2 = *(const v16h*)(Bc + 32 * BSTR);
    v16h b3 = *(const v16h*)(Bc + 48 * BSTR);
    acc[0][0] = wmma_f16(a0, b0, acc[0][0]);
    acc[1][0] = wmma_f16(a1, b0, acc[1][0]);
    acc[0][1] = wmma_f16(a0, b1, acc[0][1]);
    acc[1][1] = wmma_f16(a1, b1, acc[1][1]);
    acc[0][2] = wmma_f16(a0, b2, acc[0][2]);
    acc[1][2] = wmma_f16(a1, b2, acc[1][2]);
    acc[0][3] = wmma_f16(a0, b3, acc[0][3]);
    acc[1][3] = wmma_f16(a1, b3, acc[1][3]);
    wait_async0();
    __syncthreads();
    cur ^= 1;
  }

#pragma unroll
  for (int s = 0; s < 2; ++s) {
#pragma unroll
    for (int r = 0; r < 8; ++r) {
      int grow = m0 + s*16 + r + 8*lh;
      float* dst = out + (size_t)grow * DIMK + n0;
#pragma unroll
      for (int f = 0; f < 4; ++f) dst[f*16 + lm] = acc[s][f][r] + bo[n0 + f*16 + lm];
    }
  }
}

// ---------------- launcher ----------------

extern "C" void kernel_launch(void* const* d_in, const int* in_sizes, int n_in,
                              void* d_out, int out_size, void* d_ws, size_t ws_size,
                              hipStream_t stream) {
  (void)in_sizes; (void)n_in; (void)out_size; (void)ws_size;
  const float* x    = (const float*)d_in[0];
  const float* Wq   = (const float*)d_in[1];
  const float* Wk   = (const float*)d_in[2];
  const float* Wv   = (const float*)d_in[3];
  const float* Wo   = (const float*)d_in[4];
  const float* bo   = (const float*)d_in[5];
  const float* ln_g = (const float*)d_in[6];
  const float* ln_b = (const float*)d_in[7];
  const float* q_g  = (const float*)d_in[8];
  const float* k_g  = (const float*)d_in[9];
  float* out = (float*)d_out;

  char* ws = (char*)d_ws;
  const size_t MB = 1u << 20;
  half_t* xh   = (half_t*)(ws);              //  8 MB  [4096,1024] f16
  half_t* w16  = (half_t*)(ws +  8*MB);      //  8 MB  4 x [1024,1024] f16
  half_t* qbuf = (half_t*)(ws + 16*MB);      //  8 MB  [32,2048,64]
  half_t* kbuf = (half_t*)(ws + 24*MB);      //  8 MB  [32,2048,64]
  half_t* vT   = (half_t*)(ws + 32*MB);      //  8 MB  [32,64,2048]
  half_t* attn = (half_t*)(ws + 40*MB);      //  8 MB  [4096,1024]

  k_convert_w<<<4096, 256, 0, stream>>>(Wq, Wk, Wv, Wo, w16);
  k_layernorm<<<512, 256, 0, stream>>>(x, ln_g, ln_b, xh);
  k_qkv      <<<768, 256, 0, stream>>>(xh, w16, q_g, k_g, qbuf, kbuf, vT);
  k_attn     <<<512, 256, 0, stream>>>(qbuf, kbuf, vT, attn);
  k_outproj  <<<256, 256, 0, stream>>>(attn, w16 + 3*(size_t)DIMK*DIMK, bo, out);
}
